// Block2_4518305595593
// MI455X (gfx1250) — compile-verified, matrix operand order checked by reading, not run on hardware
//
#include <hip/hip_runtime.h>
#include <hip/hip_fp16.h>

typedef _Float16 v16h __attribute__((ext_vector_type(16)));
typedef float    v8f  __attribute__((ext_vector_type(8)));

#define HT    48
#define WT_   48
#define NPIX  2304      // 48*48
#define BATCH 2
#define PTOT  4608      // BATCH*NPIX

__device__ __forceinline__ float gelu_exact(float x) {
    return 0.5f * x * (1.0f + erff(x * 0.70710678118654752f));
}

// K-index map for 16-bit A (16x32) / B (32x16) WMMA fragments, CDNA5 ISA 7.12.2:
// lanes 0-15: VGPR j<4 -> K=2j,2j+1 ; j>=4 -> K=16+2(j-4)+... ; lanes 16-31: +8.
__device__ __forceinline__ int kmap(int j, int kbase) {
    return kbase + ((j < 4) ? (2 * j) : (8 + 2 * j));
}

// Build a 16-element f16 fragment from two contiguous 8-float groups.
__device__ __forceinline__ v16h cvt16(const float* __restrict__ p0,
                                      const float* __restrict__ p1) {
    v16h r;
    #pragma unroll
    for (int i = 0; i < 8; ++i) { r[i] = (_Float16)p0[i]; r[8 + i] = (_Float16)p1[i]; }
    return r;
}
__device__ __forceinline__ v16h cvt16s(const float* __restrict__ p0,
                                       const float* __restrict__ p1, float s) {
    v16h r;
    #pragma unroll
    for (int i = 0; i < 8; ++i) { r[i] = (_Float16)(p0[i] * s); r[8 + i] = (_Float16)(p1[i] * s); }
    return r;
}
__device__ __forceinline__ v16h zero16() {
    v16h r;
    #pragma unroll
    for (int i = 0; i < 16; ++i) r[i] = (_Float16)0.0f;
    return r;
}

// -------------------------------------------------------------------------
// Weight pre-format: fp32 [O,Cin,KH,KW] -> f16 A-fragments laid out so each
// lane's 16 fragment elements are one contiguous 32-byte block:
//   WF[(((t16*KHW + tap)*(Cin/32) + kc32)*32 + lane)*16 + e]
// -------------------------------------------------------------------------
__global__ void __launch_bounds__(256) wprep(const float* __restrict__ W,
    _Float16* __restrict__ WF, int Cout, int Cin, int KHW)
{
    const long total = (long)(Cout >> 4) * KHW * (Cin >> 5) * 32;
    const long idx = (long)blockIdx.x * 256 + threadIdx.x;
    if (idx >= total) return;
    const int lane = (int)(idx & 31);
    long rest = idx >> 5;
    const int kcn  = Cin >> 5;
    const int kc32 = (int)(rest % kcn); rest /= kcn;
    const int tap  = (int)(rest % KHW);
    const int t16  = (int)(rest / KHW);
    const int m  = lane & 15;
    const int kb = (lane >> 4) * 8;
    const int o  = t16 * 16 + m;
    const float* wp = W + (long)o * Cin * KHW + tap;
    v16h outv;
    #pragma unroll
    for (int j = 0; j < 8; ++j) {
        const int k0 = kc32 * 32 + kmap(j, kb);
        outv[2*j]   = (_Float16)wp[(long)k0 * KHW];
        outv[2*j+1] = (_Float16)wp[(long)(k0 + 1) * KHW];
    }
    *(v16h*)(WF + idx * 16) = outv;
}

// -------------------------------------------------------------------------
// Generic implicit-GEMM conv (stride 1) with fused epilogues.
// Wave: 16-position x 64-out-channel tile, 4 f32 accumulators.
// Weight A-fragments are staged once per workgroup into LDS via
// GLOBAL_LOAD_ASYNC_TO_LDS_B128 (ASYNCcnt) and served as ds_load_b128,
// removing the 8x per-wave L2 refetch of identical fragments.
// Activation layouts: xlayout/ylayout 0 = PC ([B*N][C]), 1 = CP (NCHW).
// modes: 0=raw, 1=+bias, 2=BN+GELU, 3=BN+GELU,+R,GELU, 4=+bias,+R
// grid: (36, Cout/64), block 256; dynamic LDS = min(Cin/32,16)*4096 bytes.
// -------------------------------------------------------------------------
extern __shared__ _Float16 smw[];

__global__ void __launch_bounds__(256) conv_wmma(
    const float* __restrict__ X, const _Float16* __restrict__ WF,
    const float* __restrict__ gamma, const float* __restrict__ beta,
    const float* __restrict__ bias, const float* __restrict__ R,
    float* __restrict__ Y,
    int Cin, int KH, int KW, int PH, int PW, int DH, int DW,
    int YC, int Coff, int mode, int xlayout, int ylayout)
{
    const int tid  = threadIdx.x;
    const int wid  = tid >> 5;
    const int lane = tid & 31;
    const int pt   = blockIdx.x * 8 + wid;   // 0..287
    const int ot   = blockIdx.y;             // 64-out-channel chunk

    const int m     = lane & 15;
    const int half  = lane >> 4;
    const int kbase = half * 8;

    // this lane's position for the B (activation) fragment
    const int p    = pt * 16 + m;
    const int bidx = p / NPIX;
    const int pos  = p - bidx * NPIX;
    const int ph0  = pos / WT_;
    const int pw0  = pos - ph0 * WT_;

    v8f acc[4];
    #pragma unroll
    for (int s = 0; s < 4; ++s) acc[s] = (v8f){0.f,0.f,0.f,0.f,0.f,0.f,0.f,0.f};

    const int  KHW    = KH * KW;
    const int  kcn    = Cin >> 5;                 // 8, 16 or 32 (power of two)
    const int  kchunk = (kcn < 16) ? kcn : 16;    // LDS-staged K steps per round
    const int  lg     = 31 - __clz(kchunk * 64);  // log2(chunks per subtile)
    const long tstr   = (long)KHW * kcn * 512;    // halves per 16-out tile
    const unsigned smbase = (unsigned)(size_t)(void*)smw;  // LDS byte offset

    for (int th = 0; th < KH; ++th) {
      for (int tw = 0; tw < KW; ++tw) {
        const int hh = ph0 + th * DH - PH;
        const int ww = pw0 + tw * DW - PW;
        const bool inb = (hh >= 0) && (hh < HT) && (ww >= 0) && (ww < WT_);
        const float* bpPC = X + ((long)bidx * NPIX + (long)hh * WT_ + ww) * Cin;
        const long xoffCP = (long)bidx * Cin * NPIX + (long)hh * WT_ + ww;
        const int tap = th * KW + tw;

        for (int kb0 = 0; kb0 < kcn; kb0 += kchunk) {
          // ---- async-stage A fragments (4 subtiles x kchunk steps) into LDS ----
          {
            const long tapbase = ((long)tap * kcn + kb0) * 512;  // halves
            for (int i = 0; i < kchunk; ++i) {
              const int cid = i * 256 + tid;              // 16B chunk id
              const int sub = cid >> lg;
              const int rem = cid & ((1 << lg) - 1);
              const unsigned goff = (unsigned)((((long)(ot * 4 + sub)) * tstr
                                    + tapbase + (long)rem * 8) * 2);   // bytes
              const unsigned loff = smbase + (unsigned)cid * 16;       // bytes
              asm volatile("global_load_async_to_lds_b128 %0, %1, %2"
                           :: "v"(loff), "v"(goff), "s"((const void*)WF)
                           : "memory");
            }
            asm volatile("s_wait_asynccnt 0" ::: "memory");
          }
          __syncthreads();

          for (int kcl = 0; kcl < kchunk; ++kcl) {
            const int kc = (kb0 + kcl) * 32;
            v16h bf;
            if (xlayout == 0) {
              if (inb) {
                const float* q0 = bpPC + kc + kbase;
                bf = cvt16(q0, q0 + 16);
              } else {
                bf = zero16();
              }
            } else {
              #pragma unroll
              for (int j = 0; j < 8; ++j) {
                const int k0 = kc + kmap(j, kbase);
                const float e0 = inb ? X[xoffCP + (long)k0 * NPIX]       : 0.0f;
                const float e1 = inb ? X[xoffCP + (long)(k0 + 1) * NPIX] : 0.0f;
                bf[2*j]   = (_Float16)e0;
                bf[2*j+1] = (_Float16)e1;
              }
            }
            #pragma unroll
            for (int sub = 0; sub < 4; ++sub) {
              const v16h af = *(const v16h*)(smw
                  + ((long)(sub * kchunk + kcl)) * 512 + lane * 16);
              acc[sub] = __builtin_amdgcn_wmma_f32_16x16x32_f16(
                  false, af, false, bf, (short)0, acc[sub], false, false);
            }
          }
          asm volatile("s_wait_dscnt 0" ::: "memory");  // reads done before restage
          __syncthreads();
        }
      }
    }

    // epilogue. C/D layout: VGPR r -> row o = r + 8*half (within 16), col = lane&15.
    const float invbn = 0.99999500003750f;  // 1/sqrt(1 + 1e-5)
    const int n  = lane & 15;
    const int pq = pt * 16 + n;             // global position (PC row)
    const int bo = pq / NPIX;
    const int po = pq - bo * NPIX;
    #pragma unroll
    for (int sub = 0; sub < 4; ++sub) {
      const int obase = ot * 64 + sub * 16 + 8 * half;
      float tmp[8];
      #pragma unroll
      for (int r = 0; r < 8; ++r) {
        const int o = obase + r;
        float vv = acc[sub][r];
        const long ridx = (long)pq * 256 + o;   // R: PC, 256 channels
        if (mode == 1) {
          vv += bias[o];
        } else if (mode == 2) {
          vv = gelu_exact(vv * invbn * gamma[o] + beta[o]);
        } else if (mode == 3) {
          vv = gelu_exact(vv * invbn * gamma[o] + beta[o]);
          vv = gelu_exact(vv + R[ridx]);
        } else if (mode == 4) {
          vv += bias[o] + R[ridx];
        }
        tmp[r] = vv;
      }
      if (ylayout == 0) {
        float* yp = Y + (long)pq * YC + Coff + obase;   // contiguous 8 floats
        #pragma unroll
        for (int r = 0; r < 8; ++r) yp[r] = tmp[r];
      } else {
        #pragma unroll
        for (int r = 0; r < 8; ++r)
          Y[((long)(bo * YC + Coff + obase + r)) * NPIX + po] = tmp[r];
      }
    }
}

// -------------------------------------------------------------------------
// Channel LayerNorm (C=256, eps=1e-6) on PC layout.
// -------------------------------------------------------------------------
__global__ void __launch_bounds__(256) ln2d_kernel(
    const float* __restrict__ X, const float* __restrict__ W,
    const float* __restrict__ Bv, float* __restrict__ Y)
{
    const int p = blockIdx.x * 256 + threadIdx.x;
    if (p >= PTOT) return;
    const float* xp = X + (long)p * 256;
    float s = 0.f;
    for (int c = 0; c < 256; ++c) s += xp[c];
    const float u = s * (1.0f / 256.0f);
    float v = 0.f;
    for (int c = 0; c < 256; ++c) { const float d = xp[c] - u; v += d * d; }
    const float inv = rsqrtf(v * (1.0f / 256.0f) + 1e-6f);
    float* yp = Y + (long)p * 256;
    for (int c = 0; c < 256; ++c) yp[c] = W[c] * (xp[c] - u) * inv + Bv[c];
}

// -------------------------------------------------------------------------
// Gumbel hard-selection mask (argmax over 4 logits == 0), PC-layout input.
// -------------------------------------------------------------------------
__global__ void __launch_bounds__(256) mask_kernel(
    const float* __restrict__ X, const float* __restrict__ Ws,
    const float* __restrict__ Bs, const float* __restrict__ Gum,
    int* __restrict__ Mout)
{
    const int p = blockIdx.x * 256 + threadIdx.x;
    if (p >= PTOT) return;
    const int b = p / NPIX, n = p - b * NPIX;
    const float* xp = X + (long)p * 256;
    float best = 0.f; int arg = 0;
    for (int s = 0; s < 4; ++s) {
        float l = Bs[s] + Gum[(long)(b * 4 + s) * NPIX + n];
        for (int c = 0; c < 256; ++c) l += Ws[s * 256 + c] * xp[c];
        if (s == 0) best = l;
        else if (l > best) { best = l; arg = s; }
    }
    Mout[p] = (arg == 0) ? 1 : 0;
}

// -------------------------------------------------------------------------
// Flash-style masked attention. One wave per (batch, head, 16-query tile).
// Q,K in PC layout ([B*N][256]); V in CP layout ([B,256,N]); OUT in PC.
// grid: (36, 16), block 128.
// -------------------------------------------------------------------------
__global__ void __launch_bounds__(128) attn_wmma(
    const float* __restrict__ Q, const float* __restrict__ K,
    const float* __restrict__ V, const int* __restrict__ MQ,
    const int* __restrict__ MK, float* __restrict__ OUT)
{
    __shared__ _Float16 plds[4][32 * 16];   // per-wave P tile, [key][query]
    __shared__ float    stats[4][16];       // per-wave per-query broadcast

    const int wid  = threadIdx.x >> 5;
    const int lane = threadIdx.x & 31;
    const int qt = blockIdx.x * 4 + wid;    // 0..143
    const int bh = blockIdx.y;              // 0..15
    const int h  = bh & 7;
    const int b  = bh >> 3;

    const int m     = lane & 15;
    const int half  = lane >> 4;
    const int kbase = half * 8;
    const float scale = 0.17677669529663687f;   // 32^-0.5

    // Q fragment (A layout): rows = queries, K-dim = 32 head dims, scale folded
    v16h qa;
    {
        const int qpos = qt * 16 + m;
        const float* qp = Q + ((long)(b * NPIX + qpos)) * 256 + h * 32 + kbase;
        qa = cvt16s(qp, qp + 16, scale);
    }

    v8f O0 = (v8f){0.f,0.f,0.f,0.f,0.f,0.f,0.f,0.f};
    v8f O1 = O0;
    float Mrow[8], Lrow[8];
    #pragma unroll
    for (int r = 0; r < 8; ++r) { Mrow[r] = -3.0e38f; Lrow[r] = 0.f; }

    for (int kt = 0; kt < 144; kt += 2) {
        v8f S[2];
        #pragma unroll
        for (int t = 0; t < 2; ++t) {
            const int key = (kt + t) * 16 + m;
            const float* kp = K + ((long)(b * NPIX + key)) * 256 + h * 32 + kbase;
            const v16h kf = cvt16(kp, kp + 16);     // B layout: cols = keys
            const v8f z = (v8f){0.f,0.f,0.f,0.f,0.f,0.f,0.f,0.f};
            S[t] = __builtin_amdgcn_wmma_f32_16x16x32_f16(
                false, qa, false, kf, (short)0, z, false, false);
            const bool on = MK[b * NPIX + key] != 0;  // mask by replacement
            #pragma unroll
            for (int r = 0; r < 8; ++r) S[t][r] = on ? S[t][r] : -1e9f;
        }

        // online softmax: rows = queries (r + 8*half), keys across 16 lanes
        float alpha[8];
        #pragma unroll
        for (int r = 0; r < 8; ++r) {
            float mx = fmaxf(S[0][r], S[1][r]);
            #pragma unroll
            for (int off = 1; off < 16; off <<= 1) mx = fmaxf(mx, __shfl_xor(mx, off, 32));
            const float nm = fmaxf(Mrow[r], mx);
            alpha[r] = __expf(Mrow[r] - nm);
            Mrow[r]  = nm;
            S[0][r] = __expf(S[0][r] - nm);
            S[1][r] = __expf(S[1][r] - nm);
            float sum = S[0][r] + S[1][r];
            #pragma unroll
            for (int off = 1; off < 16; off <<= 1) sum += __shfl_xor(sum, off, 32);
            Lrow[r] = Lrow[r] * alpha[r] + sum;
        }

        // broadcast per-query alpha through LDS (query index = r + 8*half)
        if (m == 0) {
            #pragma unroll
            for (int r = 0; r < 8; ++r) stats[wid][8 * half + r] = alpha[r];
        }
        // P tiles to LDS in [key][query] order
        #pragma unroll
        for (int t = 0; t < 2; ++t) {
            _Float16* bp = &plds[wid][(t * 16 + m) * 16 + 8 * half];
            #pragma unroll
            for (int r = 0; r < 8; ++r) bp[r] = (_Float16)S[t][r];
        }
        asm volatile("s_wait_dscnt 0" ::: "memory");

        const float av = stats[wid][m];          // query = lane&15 in O layout
        #pragma unroll
        for (int r = 0; r < 8; ++r) { O0[r] *= av; O1[r] *= av; }

        // P fragment (B layout): rows = 32 keys, cols = 16 queries
        v16h pf;
        #pragma unroll
        for (int j = 0; j < 8; ++j) {
            const int k0 = kmap(j, kbase);
            pf[2*j]   = plds[wid][k0 * 16 + m];
            pf[2*j+1] = plds[wid][(k0 + 1) * 16 + m];
        }
        // V fragments (A layout): rows = head dims, K-dim = 32 keys (contiguous)
        #pragma unroll
        for (int sub = 0; sub < 2; ++sub) {
            const float* vp = V + ((long)(b * 256 + h * 32 + sub * 16 + m)) * NPIX
                                + (long)kt * 16 + kbase;
            const v16h va = cvt16(vp, vp + 16);
            if (sub == 0)
                O0 = __builtin_amdgcn_wmma_f32_16x16x32_f16(false, va, false, pf, (short)0, O0, false, false);
            else
                O1 = __builtin_amdgcn_wmma_f32_16x16x32_f16(false, va, false, pf, (short)0, O1, false, false);
        }
    }

    // final 1/L per query; scatter-back (reference's where(mq, out, v)); OUT is PC.
    if (m == 0) {
        #pragma unroll
        for (int r = 0; r < 8; ++r) stats[wid][8 * half + r] = Lrow[r];
    }
    asm volatile("s_wait_dscnt 0" ::: "memory");
    const float invl = 1.0f / stats[wid][m];
    const int  qg  = qt * 16 + m;
    const bool sel = MQ[b * NPIX + qg] != 0;
    #pragma unroll
    for (int sub = 0; sub < 2; ++sub) {
        const int cbase = h * 32 + sub * 16 + 8 * half;
        float tmp[8];
        #pragma unroll
        for (int r = 0; r < 8; ++r) {
            const int c = cbase + r;
            float val = (sub == 0 ? O0[r] : O1[r]) * invl;
            if (!sel) val = V[((long)(b * 256 + c)) * NPIX + qg];
            tmp[r] = val;
        }
        float* op = OUT + ((long)(b * NPIX + qg)) * 256 + cbase;
        #pragma unroll
        for (int r = 0; r < 8; ++r) op[r] = tmp[r];
    }
}

// -------------------------------------------------------------------------
// Host side
// -------------------------------------------------------------------------
static void launch_conv(const float* X, const _Float16* WF, const float* g,
                        const float* be, const float* bias, const float* R,
                        float* Y, int Cin, int Cout, int KH, int KW,
                        int PH, int PW, int DH, int DW,
                        int YC, int Coff, int mode, int xl, int yl, hipStream_t s)
{
    const int kcn    = Cin / 32;
    const int kchunk = (kcn < 16) ? kcn : 16;
    conv_wmma<<<dim3(PTOT / 16 / 8, Cout / 64), dim3(256), kchunk * 4096, s>>>(
        X, WF, g, be, bias, R, Y, Cin, KH, KW, PH, PW, DH, DW, YC, Coff, mode, xl, yl);
}

extern "C" void kernel_launch(void* const* d_in, const int* in_sizes, int n_in,
                              void* d_out, int out_size, void* d_ws, size_t ws_size,
                              hipStream_t stream)
{
    (void)in_sizes; (void)n_in; (void)out_size; (void)ws_size;
    const float* x = (const float*)d_in[0];
    auto F = [&](int i) { return (const float*)d_in[i]; };
    const float* NUL = nullptr;

    float* ws = (float*)d_ws;
    const long S1 = (long)BATCH * 256 * NPIX;      // 1,179,648 floats
    float* cat  = ws;                               // PC [4608][1024]
    float* tmp1 = ws + 4 * S1;
    float* tmp2 = ws + 5 * S1;
    float* xr   = ws + 6 * S1;
    float* rfb  = ws + 7 * S1;
    float* ln1  = ws + 8 * S1;
    float* qb   = ws + 9 * S1;
    float* kb   = ws + 10 * S1;
    float* vb   = ws + 11 * S1;                     // CP [B,256,N]
    float* att  = ws + 12 * S1;
    float* xatt = ws + 13 * S1;
    float* ln2b = ws + 14 * S1;
    float* mlph = ws + 15 * S1;                     // PC [4608][512]
    int*   mq   = (int*)(ws + 17 * S1);
    int*   mk   = mq + PTOT;
    float* out  = (float*)d_out;

    // f16 formatted-weight arena (after masks, 64KB pad for alignment)
    _Float16* cur = (_Float16*)(ws + 17 * S1 + 16384);
    auto prep = [&](const float* W, int Cout, int Cin, int KHW) {
        _Float16* dst = cur;
        const long frag32 = (long)(Cout >> 4) * KHW * (Cin >> 5) * 32;
        wprep<<<dim3((int)((frag32 + 255) / 256)), dim3(256), 0, stream>>>(
            W, dst, Cout, Cin, KHW);
        cur += frag32 * 16;
        return (const _Float16*)dst;
    };

    // ---- pre-format all conv weights ----
    const _Float16* wb0  = prep(F(1),  256, 256, 1);
    const _Float16* wb10 = prep(F(4),  256, 256, 1);
    const _Float16* wb11 = prep(F(7),  256, 256, 3);
    const _Float16* wb12 = prep(F(10), 256, 256, 3);
    const _Float16* wb13 = prep(F(13), 256, 256, 9);
    const _Float16* wb20 = prep(F(16), 256, 256, 1);
    const _Float16* wb21 = prep(F(19), 256, 256, 5);
    const _Float16* wb22 = prep(F(22), 256, 256, 5);
    const _Float16* wb23 = prep(F(25), 256, 256, 9);
    const _Float16* wb30 = prep(F(28), 256, 256, 1);
    const _Float16* wb31 = prep(F(31), 256, 256, 7);
    const _Float16* wb32 = prep(F(34), 256, 256, 7);
    const _Float16* wb33 = prep(F(37), 256, 256, 9);
    const _Float16* wcat = prep(F(40), 256, 1024, 9);
    const _Float16* wres = prep(F(43), 256, 256, 1);
    const _Float16* wq_  = prep(F(50), 256, 256, 1);
    const _Float16* wk_  = prep(F(52), 256, 256, 1);
    const _Float16* wv_  = prep(F(54), 256, 256, 1);
    const _Float16* wp_  = prep(F(56), 256, 256, 1);
    const _Float16* wm1  = prep(F(62), 512, 256, 1);
    const _Float16* wm2  = prep(F(64), 256, 512, 1);

    // ---------------- RFB ----------------
    launch_conv(x, wb0, F(2), F(3), NUL, NUL, cat, 256,256, 1,1, 0,0, 1,1, 1024,0, 2, 1,0, stream);
    // branch 1: 1x1, 1x3(p0,1), 3x1(p1,0), 3x3 d3 p3 -> cat[:,256:512]
    launch_conv(x,    wb10, F(5),  F(6),  NUL, NUL, tmp1, 256,256, 1,1, 0,0, 1,1, 256,0, 2, 1,0, stream);
    launch_conv(tmp1, wb11, F(8),  F(9),  NUL, NUL, tmp2, 256,256, 1,3, 0,1, 1,1, 256,0, 2, 0,0, stream);
    launch_conv(tmp2, wb12, F(11), F(12), NUL, NUL, tmp1, 256,256, 3,1, 1,0, 1,1, 256,0, 2, 0,0, stream);
    launch_conv(tmp1, wb13, F(14), F(15), NUL, NUL, cat,  256,256, 3,3, 3,3, 3,3, 1024,256, 2, 0,0, stream);
    // branch 2: 1x1, 1x5(p0,2), 5x1(p2,0), 3x3 d5 p5 -> cat[:,512:768]
    launch_conv(x,    wb20, F(17), F(18), NUL, NUL, tmp1, 256,256, 1,1, 0,0, 1,1, 256,0, 2, 1,0, stream);
    launch_conv(tmp1, wb21, F(20), F(21), NUL, NUL, tmp2, 256,256, 1,5, 0,2, 1,1, 256,0, 2, 0,0, stream);
    launch_conv(tmp2, wb22, F(23), F(24), NUL, NUL, tmp1, 256,256, 5,1, 2,0, 1,1, 256,0, 2, 0,0, stream);
    launch_conv(tmp1, wb23, F(26), F(27), NUL, NUL, cat,  256,256, 3,3, 5,5, 5,5, 1024,512, 2, 0,0, stream);
    // branch 3: 1x1, 1x7(p0,3), 7x1(p3,0), 3x3 d7 p7 -> cat[:,768:1024]
    launch_conv(x,    wb30, F(29), F(30), NUL, NUL, tmp1, 256,256, 1,1, 0,0, 1,1, 256,0, 2, 1,0, stream);
    launch_conv(tmp1, wb31, F(32), F(33), NUL, NUL, tmp2, 256,256, 1,7, 0,3, 1,1, 256,0, 2, 0,0, stream);
    launch_conv(tmp2, wb32, F(35), F(36), NUL, NUL, tmp1, 256,256, 7,1, 3,0, 1,1, 256,0, 2, 0,0, stream);
    launch_conv(tmp1, wb33, F(38), F(39), NUL, NUL, cat,  256,256, 3,3, 7,7, 7,7, 1024,768, 2, 0,0, stream);
    // residual branch
    launch_conv(x, wres, F(44), F(45), NUL, NUL, xr, 256,256, 1,1, 0,0, 1,1, 256,0, 2, 1,0, stream);
    // cat conv (Cin=1024, 3x3 p1) fused with gelu(xc + xr)
    launch_conv(cat, wcat, F(41), F(42), NUL, xr, rfb, 1024,256, 3,3, 1,1, 1,1, 256,0, 3, 0,0, stream);

    // ---------------- LN1 + masks ----------------
    ln2d_kernel<<<dim3(PTOT / 256), dim3(256), 0, stream>>>(rfb, F(58), F(59), ln1);
    mask_kernel<<<dim3(PTOT / 256), dim3(256), 0, stream>>>(ln1, F(46), F(47), F(66), mq);
    mask_kernel<<<dim3(PTOT / 256), dim3(256), 0, stream>>>(ln1, F(48), F(49), F(67), mk);

    // ---------------- Attention ----------------
    launch_conv(ln1, wq_, NUL, NUL, F(51), NUL, qb, 256,256, 1,1, 0,0, 1,1, 256,0, 1, 0,0, stream);
    launch_conv(ln1, wk_, NUL, NUL, F(53), NUL, kb, 256,256, 1,1, 0,0, 1,1, 256,0, 1, 0,0, stream);
    launch_conv(ln1, wv_, NUL, NUL, F(55), NUL, vb, 256,256, 1,1, 0,0, 1,1, 256,0, 1, 0,1, stream);
    attn_wmma<<<dim3(36, 16), dim3(128), 0, stream>>>(qb, kb, vb, mq, mk, att);
    // projection + shortcut (sc = rfb)
    launch_conv(att, wp_, NUL, NUL, F(57), rfb, xatt, 256,256, 1,1, 0,0, 1,1, 256,0, 4, 0,0, stream);

    // ---------------- LN2 + MLP + residual ----------------
    ln2d_kernel<<<dim3(PTOT / 256), dim3(256), 0, stream>>>(xatt, F(60), F(61), ln2b);
    launch_conv(ln2b, wm1, NUL, NUL, F(63), NUL,  mlph, 256,512, 1,1, 0,0, 1,1, 512,0, 1, 0,0, stream);
    // final: CP (NCHW) output, residual xatt (PC)
    launch_conv(mlph, wm2, NUL, NUL, F(65), xatt, out,  512,256, 1,1, 0,0, 1,1, 256,0, 4, 0,1, stream);
}
